// Attention_25228637897408
// MI455X (gfx1250) — matrix operand order repaired and checked by reading, not compiled, not run on hardware
//
#include <hip/hip_runtime.h>
#include <hip/hip_bf16.h>
#include <math.h>

// ---------------------------------------------------------------------------
// MI455X / gfx1250 fused attention block (wave32, WMMA bf16 -> f32)
//   q = rmsnorm(x @ wq^T) -> rope
//   k = rmsnorm(((1-g)x + g*skip) @ wk^T) -> rope      (g = sigmoid(0.1*ln_s))
//   v = ((1-g)x + g*skip) @ wv^T + (1-g)*v_bias
//   y = softmax(causal(q k^T * gain^2/sqrt(HD))) v
//   out = (y @ wproj^T) * ln_s
// ---------------------------------------------------------------------------

typedef __attribute__((ext_vector_type(16))) __bf16       bf16x16;
typedef __attribute__((ext_vector_type(8)))  float        f32x8;
typedef __attribute__((ext_vector_type(4)))  unsigned int u32x4;
typedef int gvi32x4 __attribute__((vector_size(16)));  // GCC vector: matches builtin pointee

union FragU { bf16x16 v; u32x4 q[2]; };

#define WMMA_BF16(A, B, C) \
  __builtin_amdgcn_wmma_f32_16x16x32_bf16(false, (A), false, (B), (short)0, (C), false, false)

#if __has_builtin(__builtin_amdgcn_global_load_async_to_lds_b128) && \
    __has_builtin(__builtin_amdgcn_s_wait_asynccnt)
#define USE_ASYNC_LDS 1
#else
#define USE_ASYNC_LDS 0
#endif

// 16-byte global -> LDS copy (async on CDNA5 path, sync fallback otherwise)
__device__ __forceinline__ void cp16_g2l(const __bf16* g, __bf16* l) {
#if USE_ASYNC_LDS
  __builtin_amdgcn_global_load_async_to_lds_b128(
      (__attribute__((address_space(1))) gvi32x4*)g,
      (__attribute__((address_space(3))) gvi32x4*)l, 0, 0);
#else
  *(u32x4*)l = *(const u32x4*)g;
#endif
}

static constexpr int Tn  = 2048;
static constexpr int Dn  = 2048;
static constexpr int DKV = 512;
static constexpr int NH  = 16;
static constexpr int HD  = 128;
static constexpr int LDA = 40;  // LDS row stride in bf16 (80B: 16B-aligned, low-conflict)

// ---------------------------------------------------------------------------
// Elementwise prep: bf16 conversion + skip mixing
// ---------------------------------------------------------------------------
__global__ __launch_bounds__(256) void mix_kernel(const float* __restrict__ x,
                                                  const float* __restrict__ skip,
                                                  const float* __restrict__ ln_s,
                                                  __bf16* __restrict__ xb,
                                                  __bf16* __restrict__ xm, int n) {
  float g = 1.0f / (1.0f + __expf(-0.1f * ln_s[0]));
  int i = blockIdx.x * blockDim.x + threadIdx.x;
  if (i < n) {
    float xv = x[i];
    xb[i] = (__bf16)xv;
    xm[i] = (__bf16)((1.0f - g) * xv + g * skip[i]);
  }
}

__global__ __launch_bounds__(256) void cvt_kernel(const float* __restrict__ s,
                                                  __bf16* __restrict__ d, int n) {
  int i = blockIdx.x * blockDim.x + threadIdx.x;
  if (i < n) d[i] = (__bf16)s[i];
}

// ---------------------------------------------------------------------------
// bf16 WMMA GEMM:  C[M,N] = A[M,K] @ W[N,K]^T  (f32 accum)
// Block tile 128x128, 8 waves, each wave 64x32 (4 M-sub x 2 N-sub of 16x16).
// Main loop: double-buffered LDS stages (BK=32) filled with async-to-LDS
// copies; fragments read from LDS with ds_load_b128.
// Epilogue modes: 0=Q (rmsnorm+rope -> [h][t][hd]), 1=K (same -> [kv][t][hd]),
//                 2=V (+ (1-g)*bias -> transposed [kv][hd][t]), 3=proj (*ln_s f32)
// ---------------------------------------------------------------------------
template <int MODE>
__global__ __launch_bounds__(256) void gemm_kernel(const __bf16* __restrict__ A,
                                                   const __bf16* __restrict__ W,
                                                   const float* __restrict__ bias,
                                                   const float* __restrict__ ln_s,
                                                   __bf16* __restrict__ obf,
                                                   float* __restrict__ of32,
                                                   int M, int N, int K, int T) {
  __shared__ __align__(16) union SmemT {
    __bf16 stg[2][2][128 * LDA];  // [buf][0=A,1=B][row*LDA + k]
    float  ctile[64 * 129];       // epilogue staging (reused after main loop)
  } sm;

  const int tid  = threadIdx.x;
  const int lane = tid & 31;
  const int wid  = tid >> 5;
  const int hl   = lane >> 4;  // half-wave select
  const int l15  = lane & 15;
  const int bm = blockIdx.y * 128, bn = blockIdx.x * 128;
  const int wm = (wid >> 2) * 64, wn = (wid & 3) * 32;

  // Per-thread staging assignment: 4 x 16B chunks (2 for A-tile, 2 for B-tile).
  // Chunk id c in [0,512): row = c>>2, 16B-chunk-in-row = c&3.
  const __bf16* gsrc[4];
  int loff[4];
#pragma unroll
  for (int i = 0; i < 2; ++i) {
    int c = i * 256 + tid;
    int row = c >> 2, cir = c & 3;
    gsrc[i]     = A + (size_t)(bm + row) * K + cir * 8;
    gsrc[i + 2] = W + (size_t)(bn + row) * K + cir * 8;
    loff[i] = loff[i + 2] = row * LDA + cir * 8;
  }
  auto issue = [&](int buf, int k0) {
#pragma unroll
    for (int i = 0; i < 2; ++i) cp16_g2l(gsrc[i] + k0, &sm.stg[buf][0][loff[i]]);
#pragma unroll
    for (int i = 0; i < 2; ++i) cp16_g2l(gsrc[i + 2] + k0, &sm.stg[buf][1][loff[i + 2]]);
  };

  f32x8 acc[4][2] = {};
  issue(0, 0);
  int buf = 0;
  for (int k0 = 0; k0 < K; k0 += 32) {
    const bool nxt = (k0 + 32) < K;
    if (nxt) issue(buf ^ 1, k0 + 32);
#if USE_ASYNC_LDS
    if (nxt) __builtin_amdgcn_s_wait_asynccnt(4);  // stage n done (in-order)
    else     __builtin_amdgcn_s_wait_asynccnt(0);
#endif
    __syncthreads();  // all waves' copies visible

    const __bf16* sa = sm.stg[buf][0];
    const __bf16* sb = sm.stg[buf][1];
    FragU af[4], bfr[2];
#pragma unroll
    for (int i = 0; i < 4; ++i) {
      const __bf16* r = sa + (wm + i * 16 + l15) * LDA;
      af[i].q[0] = *(const u32x4*)(r + hl * 8);
      af[i].q[1] = *(const u32x4*)(r + 16 + hl * 8);
    }
#pragma unroll
    for (int j = 0; j < 2; ++j) {
      const __bf16* r = sb + (wn + j * 16 + l15) * LDA;
      bfr[j].q[0] = *(const u32x4*)(r + hl * 8);
      bfr[j].q[1] = *(const u32x4*)(r + 16 + hl * 8);
    }
#pragma unroll
    for (int i = 0; i < 4; ++i)
#pragma unroll
      for (int j = 0; j < 2; ++j) acc[i][j] = WMMA_BF16(af[i].v, bfr[j].v, acc[i][j]);

    __syncthreads();  // done reading buf before it is refilled
    buf ^= 1;
  }

  // ------------------------- epilogue (two 64-row phases) -------------------
  const float invHD = 1.0f / 128.0f;
  for (int p = 0; p < 2; ++p) {
    __syncthreads();
    if (wm == p * 64) {
#pragma unroll
      for (int i = 0; i < 4; ++i)
#pragma unroll
        for (int j = 0; j < 2; ++j)
#pragma unroll
          for (int v = 0; v < 8; ++v)
            sm.ctile[(i * 16 + v + 8 * hl) * 129 + wn + j * 16 + l15] = acc[i][j][v];
    }
    __syncthreads();

    if (MODE == 0 || MODE == 1) {
      // rmsnorm over head dim (tile N==128==one head) + rope; 2 threads/row
      if (threadIdx.x < 128) {
        int r = threadIdx.x >> 1, sel = threadIdx.x & 1;
        int grow = bm + p * 64 + r;  // token index
        const float* crow = &sm.ctile[r * 129];
        float ssq = 0.f;
#pragma unroll 8
        for (int jj = 0; jj < 32; ++jj) {
          float a = crow[sel * 32 + jj];
          float b = crow[sel * 32 + jj + 64];
          ssq += a * a + b * b;
        }
        ssq += __shfl_xor(ssq, 1, 32);  // partner thread covers other half of row
        float rn = rsqrtf(ssq * invHD + 1.1920929e-7f);
        float tpos = (float)grow;
        int hh = bn >> 7;  // head (Q) or kv-head (K)
        __bf16* orow = obf + ((size_t)hh * T + grow) * HD;
#pragma unroll 4
        for (int jj = 0; jj < 32; ++jj) {
          int d = sel * 32 + jj;
          // inv_freq = 10000^(-2d/128) = exp(-d * ln(1e4)/64)
          float fr = tpos * __expf((float)d * -0.14391156831f);
          float sn, cs;
          __sincosf(fr, &sn, &cs);
          float x1 = crow[d] * rn, x2 = crow[d + 64] * rn;
          orow[d]      = (__bf16)(x1 * cs + x2 * sn);
          orow[d + 64] = (__bf16)(x2 * cs - x1 * sn);
        }
      }
    } else if (MODE == 2) {
      float g = 1.0f / (1.0f + __expf(-0.1f * ln_s[0]));
      int kvh = bn >> 7;
      for (int idx = threadIdx.x; idx < 64 * 128; idx += 256) {
        int r = idx >> 7, c = idx & 127;
        float val = sm.ctile[r * 129 + c] + (1.0f - g) * bias[bn + c];
        // transposed store: Vt[kv][d][t]  (makes attention B-fragments contiguous)
        obf[(size_t)kvh * HD * T + (size_t)c * T + (bm + p * 64 + r)] = (__bf16)val;
      }
    } else {
      float sc = ln_s[0];
      for (int idx = threadIdx.x; idx < 64 * 128; idx += 256) {
        int r = idx >> 7, c = idx & 127;
        of32[(size_t)(bm + p * 64 + r) * N + bn + c] = sm.ctile[r * 129 + c] * sc;
      }
    }
  }
}

// ---------------------------------------------------------------------------
// Causal GQA flash attention. Block = 128 q rows of one head; each wave owns
// 16 q rows, streams 32-key tiles (per-wave causal trip counts). QK^T and P.V
// via WMMA; P bounced through a per-wave LDS tile (DS pipe is in-order per
// wave) to reshape C->A layout. Per-head |qk_g|^2 folded into the logit scale.
// ---------------------------------------------------------------------------
__global__ __launch_bounds__(256) void flash_kernel(const __bf16* __restrict__ Qb,
                                                    const __bf16* __restrict__ Kb,
                                                    const __bf16* __restrict__ Vtb,
                                                    const float* __restrict__ qkg,
                                                    __bf16* __restrict__ Yb, int T) {
  __shared__ __align__(16) __bf16 pbuf[8 * 16 * LDA];  // per-wave 16x32 (stride 40)

  const int lane = threadIdx.x & 31;
  const int wid  = threadIdx.x >> 5;
  const int hl   = lane >> 4;
  const int l15  = lane & 15;
  const int h    = blockIdx.y;
  const int kv   = h >> 2;  // REP = NH/NKV = 4
  const int qbase = blockIdx.x * 128 + wid * 16;

  float gg = qkg[h];
  float sc = gg * gg * 0.08838834764831845f;  // gain^2 / sqrt(128)

  // Q fragments for the 4 K-dim (HD) steps, resident for whole kernel
  const __bf16* qrow = Qb + ((size_t)h * T + qbase + l15) * HD;
  FragU qf[4];
#pragma unroll
  for (int kk = 0; kk < 4; ++kk) {
    qf[kk].q[0] = *(const u32x4*)(qrow + kk * 32 + hl * 8);
    qf[kk].q[1] = *(const u32x4*)(qrow + kk * 32 + 16 + hl * 8);
  }

  f32x8 o[8] = {};
  float mi[8], li[8];
#pragma unroll
  for (int v = 0; v < 8; ++v) { mi[v] = -1e30f; li[v] = 0.f; }

  const __bf16* kbp = Kb + (size_t)kv * T * HD;
  const __bf16* vbp = Vtb + (size_t)kv * HD * T;
  const int wlds = wid * 16 * LDA;

  const int nkt = (qbase + 16 + 31) >> 5;  // causal: key tiles needed by this wave
  for (int j = 0; j < nkt; ++j) {
    const int kb = j * 32;

    // ---- S = Q @ K^T (16 x 32) --------------------------------------------
    f32x8 s[2] = {};
#pragma unroll
    for (int n = 0; n < 2; ++n) {
      const __bf16* krow = kbp + (size_t)(kb + n * 16 + l15) * HD;
#pragma unroll
      for (int kk = 0; kk < 4; ++kk) {
        FragU kf;
        kf.q[0] = *(const u32x4*)(krow + kk * 32 + hl * 8);
        kf.q[1] = *(const u32x4*)(krow + kk * 32 + 16 + hl * 8);
        s[n] = WMMA_BF16(qf[kk].v, kf.v, s[n]);
      }
    }

    // ---- online softmax ----------------------------------------------------
    float p0a[8], p1a[8], rm[8];
#pragma unroll
    for (int v = 0; v < 8; ++v) {
      int row = qbase + v + 8 * hl;
      float s0 = s[0][v] * sc;
      float s1 = s[1][v] * sc;
      if (kb + l15 > row)      s0 = -1e30f;
      if (kb + 16 + l15 > row) s1 = -1e30f;
      p0a[v] = s0; p1a[v] = s1;
      float r = fmaxf(s0, s1);
      r = fmaxf(r, __shfl_xor(r, 1, 32));
      r = fmaxf(r, __shfl_xor(r, 2, 32));
      r = fmaxf(r, __shfl_xor(r, 4, 32));
      r = fmaxf(r, __shfl_xor(r, 8, 32));
      rm[v] = r;
    }
#pragma unroll
    for (int v = 0; v < 8; ++v) {
      float mnew  = fmaxf(mi[v], rm[v]);
      float alpha = __expf(mi[v] - mnew);
      float p0 = __expf(p0a[v] - mnew);
      float p1 = __expf(p1a[v] - mnew);
      float rs = p0 + p1;
      rs += __shfl_xor(rs, 1, 32);
      rs += __shfl_xor(rs, 2, 32);
      rs += __shfl_xor(rs, 4, 32);
      rs += __shfl_xor(rs, 8, 32);
      li[v] = li[v] * alpha + rs;
      mi[v] = mnew;
#pragma unroll
      for (int n = 0; n < 8; ++n) o[n][v] *= alpha;
      int prow = v + 8 * hl;
      pbuf[wlds + prow * LDA + l15]      = (__bf16)p0;
      pbuf[wlds + prow * LDA + 16 + l15] = (__bf16)p1;
    }
    asm volatile("" ::: "memory");  // keep DS store->load order (DS pipe in-order/wave)

    // reload P as A-fragment (16x32)
    FragU pf;
    pf.q[0] = *(const u32x4*)&pbuf[wlds + l15 * LDA + hl * 8];
    pf.q[1] = *(const u32x4*)&pbuf[wlds + l15 * LDA + 16 + hl * 8];

    // ---- O += P @ V (V transposed -> contiguous B-fragments) --------------
#pragma unroll
    for (int n = 0; n < 8; ++n) {
      const __bf16* vrow = vbp + (size_t)(n * 16 + l15) * T + kb + hl * 8;
      FragU vf;
      vf.q[0] = *(const u32x4*)(vrow);
      vf.q[1] = *(const u32x4*)(vrow + 16);
      o[n] = WMMA_BF16(pf.v, vf.v, o[n]);
    }
  }

  // ---- normalize + store y[t][h*128+d] as bf16 -----------------------------
#pragma unroll
  for (int v = 0; v < 8; ++v) {
    float inv = 1.0f / li[v];
    int t = qbase + v + 8 * hl;
    __bf16* yrow = Yb + (size_t)t * Dn + h * HD;
#pragma unroll
    for (int n = 0; n < 8; ++n) yrow[n * 16 + l15] = (__bf16)(o[n][v] * inv);
  }
}

// ---------------------------------------------------------------------------
extern "C" void kernel_launch(void* const* d_in, const int* in_sizes, int n_in,
                              void* d_out, int out_size, void* d_ws, size_t ws_size,
                              hipStream_t stream) {
  const float* x     = (const float*)d_in[0];
  const float* skip  = (const float*)d_in[1];
  const float* wq    = (const float*)d_in[2];
  const float* wk    = (const float*)d_in[3];
  const float* wv    = (const float*)d_in[4];
  const float* wproj = (const float*)d_in[5];
  const float* qkg   = (const float*)d_in[6];
  const float* ln_s  = (const float*)d_in[7];
  const float* vbias = (const float*)d_in[8];
  float* out = (float*)d_out;

  char* ws = (char*)d_ws;
  size_t off = 0;
  auto alloc = [&](size_t bytes) -> void* {
    void* p = ws + off;
    off += (bytes + 255) & ~(size_t)255;
    return p;
  };
  __bf16* xb  = (__bf16*)alloc((size_t)Tn * Dn * 2);
  __bf16* xm  = (__bf16*)alloc((size_t)Tn * Dn * 2);
  __bf16* wqb = (__bf16*)alloc((size_t)Dn * Dn * 2);
  __bf16* wkb = (__bf16*)alloc((size_t)DKV * Dn * 2);
  __bf16* wvb = (__bf16*)alloc((size_t)DKV * Dn * 2);
  __bf16* wpb = (__bf16*)alloc((size_t)Dn * Dn * 2);
  __bf16* qb  = (__bf16*)alloc((size_t)NH * Tn * HD * 2);
  __bf16* kbf = (__bf16*)alloc((size_t)4 * Tn * HD * 2);
  __bf16* vtb = (__bf16*)alloc((size_t)4 * HD * Tn * 2);
  __bf16* yb  = (__bf16*)alloc((size_t)Tn * Dn * 2);

  const int n_xd = Tn * Dn;
  mix_kernel<<<(n_xd + 255) / 256, 256, 0, stream>>>(x, skip, ln_s, xb, xm, n_xd);
  cvt_kernel<<<(Dn * Dn + 255) / 256, 256, 0, stream>>>(wq, wqb, Dn * Dn);
  cvt_kernel<<<(DKV * Dn + 255) / 256, 256, 0, stream>>>(wk, wkb, DKV * Dn);
  cvt_kernel<<<(DKV * Dn + 255) / 256, 256, 0, stream>>>(wv, wvb, DKV * Dn);
  cvt_kernel<<<(Dn * Dn + 255) / 256, 256, 0, stream>>>(wproj, wpb, Dn * Dn);

  // Q = rmsnorm(x @ wq^T) + rope        -> [h][t][hd]
  gemm_kernel<0><<<dim3(Dn / 128, Tn / 128), 256, 0, stream>>>(
      xb, wqb, nullptr, ln_s, qb, nullptr, Tn, Dn, Dn, Tn);
  // K = rmsnorm(xmix @ wk^T) + rope     -> [kv][t][hd]
  gemm_kernel<1><<<dim3(DKV / 128, Tn / 128), 256, 0, stream>>>(
      xm, wkb, nullptr, ln_s, kbf, nullptr, Tn, DKV, Dn, Tn);
  // V = xmix @ wv^T + (1-g)*bias        -> transposed [kv][hd][t]
  gemm_kernel<2><<<dim3(DKV / 128, Tn / 128), 256, 0, stream>>>(
      xm, wvb, vbias, ln_s, vtb, nullptr, Tn, DKV, Dn, Tn);
  // y = attention(Q, K, V)              -> [t][h*hd]
  flash_kernel<<<dim3(Tn / 128, NH), 256, 0, stream>>>(qb, kbf, vtb, qkg, yb, Tn);
  // out = (y @ wproj^T) * ln_s          -> f32
  gemm_kernel<3><<<dim3(Dn / 128, Tn / 128), 256, 0, stream>>>(
      yb, wpb, nullptr, ln_s, nullptr, out, Tn, Dn, Dn, Tn);
}
